// MessageFunction_30107720745104
// MI455X (gfx1250) — matrix operand order repaired and checked by reading, not compiled, not run on hardware
//
#include <hip/hip_runtime.h>

// ---------------------------------------------------------------------------
// CFConv (SchNet) fused edge-message kernel for MI455X (gfx1250, wave32).
//
//  W   = ssp(edge_attr @ w1^T + b1); W = ssp(W @ w2^T + b2)   [E,128]
//  msg = (h[src] @ fc1^T) * W                                  [E,128]
//  out = segment_sum(msg, dst)                                 [N,128]
//
// Strategy: per wave, process 32 edges (2 WMMA m-tiles). All three GEMMs use
// v_wmma_f32_16x16x32_f16 (fp32 accum). Weights are converted fp32->f16 into
// LDS once per block. Stage-1 output is bounced through a per-wave LDS tile
// to convert C-layout -> A-layout for the chained stage-2 GEMM. Scatter-add
// uses global fp32 atomics (out table is L2-resident: 25.6 MB << 192 MB).
// ---------------------------------------------------------------------------

typedef __attribute__((ext_vector_type(16))) _Float16 v16h;
typedef __attribute__((ext_vector_type(8)))  _Float16 v8h;
typedef __attribute__((ext_vector_type(8)))  float    v8f;
typedef __attribute__((ext_vector_type(4)))  float    f4;

#define HDIM 128
#define FDIM 128
#define WAVES 8
#define MT 2            // m-tiles (16 rows each) per wave iteration -> 32 edges

union FragH { v16h v; v8h h2[2]; _Float16 h[16]; };

__device__ __forceinline__ float ssp(float x) {
  // softplus(x) - log(2), numerically stable; __expf/__logf -> v_exp/v_log
  float ax = __builtin_fabsf(x);
  return fmaxf(x, 0.0f) + __logf(1.0f + __expf(-ax)) - 0.6931471805599453f;
}

// A fragment (16x32 f16) from a row-major fp32 row (ld=128), converted to f16.
// Lane holds row m=lane&15; chunks at k = kbase + 8*sel and +16 (ISA 7.12.2).
__device__ __forceinline__ v16h load_a_f32(const float* __restrict__ row,
                                           int kbase, int sel) {
  const f4* p = (const f4*)(row + kbase + sel * 8);
  const f4* q = (const f4*)(row + kbase + sel * 8 + 16);
  f4 a0 = p[0], a1 = p[1], c0 = q[0], c1 = q[1];
  FragH r;
  r.h[0]  = (_Float16)a0.x; r.h[1]  = (_Float16)a0.y;
  r.h[2]  = (_Float16)a0.z; r.h[3]  = (_Float16)a0.w;
  r.h[4]  = (_Float16)a1.x; r.h[5]  = (_Float16)a1.y;
  r.h[6]  = (_Float16)a1.z; r.h[7]  = (_Float16)a1.w;
  r.h[8]  = (_Float16)c0.x; r.h[9]  = (_Float16)c0.y;
  r.h[10] = (_Float16)c0.z; r.h[11] = (_Float16)c0.w;
  r.h[12] = (_Float16)c1.x; r.h[13] = (_Float16)c1.y;
  r.h[14] = (_Float16)c1.z; r.h[15] = (_Float16)c1.w;
  return r.v;
}

// A fragment from a row-major f16 LDS tile (ld=128).
__device__ __forceinline__ v16h load_a_lds(const _Float16* t1row, int kc, int sel) {
  const _Float16* p = t1row + kc * 32 + sel * 8;
  FragH r;
  r.h2[0] = *(const v8h*)p;
  r.h2[1] = *(const v8h*)(p + 16);
  return r.v;
}

// B fragment (32x16 f16) for C = A @ w^T: B[k][n] = w[ftile*16+n][k].
// Lane holds column n=lane&15; chunks at k = kc*32 + 8*sel and +16 (ISA 7.12.4).
__device__ __forceinline__ v16h load_b_lds(const _Float16* w, int ftile, int kc,
                                           int n, int sel) {
  const _Float16* p = w + (ftile * 16 + n) * HDIM + kc * 32 + sel * 8;
  FragH r;
  r.h2[0] = *(const v8h*)p;
  r.h2[1] = *(const v8h*)(p + 16);
  return r.v;
}

__global__ void zero_kernel(float* __restrict__ p, int n4) {
  int i = blockIdx.x * blockDim.x + threadIdx.x;
  if (i < n4) ((f4*)p)[i] = (f4){0.f, 0.f, 0.f, 0.f};
}

__global__ __launch_bounds__(32 * WAVES) void cfconv_kernel(
    const float* __restrict__ hfeat,   // [N,128]
    const int*   __restrict__ ei,      // [2,E] (int32)
    const float* __restrict__ ea,      // [E,128]
    const float* __restrict__ fc1,     // [128,128]
    const float* __restrict__ w1,      // [128,128]
    const float* __restrict__ b1,      // [128]
    const float* __restrict__ w2,      // [128,128]
    const float* __restrict__ b2,      // [128]
    float* __restrict__ out,           // [N,128]
    int E) {
  extern __shared__ __align__(32) _Float16 lds[];
  _Float16* sW1 = lds;                 // 16384 halves
  _Float16* sW2 = sW1 + HDIM * FDIM;
  _Float16* sFC = sW2 + HDIM * FDIM;
  float*    sB1 = (float*)(sFC + HDIM * FDIM);
  float*    sB2 = sB1 + FDIM;
  _Float16* sT1base = (_Float16*)(sB2 + FDIM);   // WAVES * MT*16*128 halves

  const int tid = threadIdx.x;
  // cooperative fp32 -> f16 weight staging (weights are L2-resident)
  for (int i = tid; i < HDIM * FDIM; i += blockDim.x) {
    sW1[i] = (_Float16)w1[i];
    sW2[i] = (_Float16)w2[i];
    sFC[i] = (_Float16)fc1[i];
  }
  if (tid < FDIM) { sB1[tid] = b1[tid]; sB2[tid] = b2[tid]; }
  __syncthreads();

  const int lane = tid & 31;
  const int wave = tid >> 5;
  const int n    = lane & 15;
  const int sel  = lane >> 4;
  _Float16* sT1 = sT1base + wave * (MT * 16 * HDIM);

  const int* srcIdx = ei;
  const int* dstIdx = ei + E;
  const int nTiles = (E + MT * 16 - 1) / (MT * 16);
  const int gwave  = blockIdx.x * (blockDim.x >> 5) + wave;
  const int nwaves = gridDim.x * (blockDim.x >> 5);

  for (int tile = gwave; tile < nTiles; tile += nwaves) {
    const int e0 = tile * (MT * 16);

    // ---- A fragments of edge_attr (fp32 -> f16) ----
    v16h Aea[MT][4];
    int erow[MT];
    #pragma unroll
    for (int mt = 0; mt < MT; ++mt) {
      int e = e0 + mt * 16 + n;
      e = e < E ? e : E - 1;
      erow[mt] = e;
      const float* row = ea + (size_t)e * HDIM;
      #pragma unroll
      for (int kc = 0; kc < 4; ++kc) Aea[mt][kc] = load_a_f32(row, kc * 32, sel);
    }

    // ---- stage 1: T1 = ssp(ea @ w1^T + b1) -> LDS (f16, A-layout source) ----
    for (int t = 0; t < 8; ++t) {
      v8f acc[MT] = {};
      #pragma unroll
      for (int kc = 0; kc < 4; ++kc) {
        v16h bfrag = load_b_lds(sW1, t, kc, n, sel);
        #pragma unroll
        for (int mt = 0; mt < MT; ++mt)
          acc[mt] = __builtin_amdgcn_wmma_f32_16x16x32_f16(
              false, Aea[mt][kc], false, bfrag, (short)0, acc[mt], false, false);
      }
      float bias = sB1[t * 16 + n];
      #pragma unroll
      for (int mt = 0; mt < MT; ++mt)
        #pragma unroll
        for (int r = 0; r < 8; ++r) {
          float v = ssp(acc[mt][r] + bias);
          // C element (m = r+8*sel, f = t*16+n)
          sT1[(mt * 16 + r + 8 * sel) * HDIM + t * 16 + n] = (_Float16)v;
        }
    }
    asm volatile("s_wait_dscnt 0" ::: "memory");  // wave-local LDS RAW

    // ---- reload T1 as A fragments ----
    v16h At1[MT][4];
    #pragma unroll
    for (int mt = 0; mt < MT; ++mt) {
      const _Float16* t1row = sT1 + (mt * 16 + n) * HDIM;
      #pragma unroll
      for (int kc = 0; kc < 4; ++kc) At1[mt][kc] = load_a_lds(t1row, kc, sel);
    }

    // ---- gather h[src] as A fragments (fp32 -> f16) ----
    v16h Axj[MT][4];
    #pragma unroll
    for (int mt = 0; mt < MT; ++mt) {
      int s = srcIdx[erow[mt]];
      const float* row = hfeat + (size_t)s * HDIM;
      #pragma unroll
      for (int kc = 0; kc < 4; ++kc) Axj[mt][kc] = load_a_f32(row, kc * 32, sel);
    }

    // ---- destination offsets + tail masks for this lane's C rows ----
    int   obase[MT][8];
    float vmask[MT][8];
    #pragma unroll
    for (int mt = 0; mt < MT; ++mt)
      #pragma unroll
      for (int r = 0; r < 8; ++r) {
        int e = e0 + mt * 16 + r + 8 * sel;
        vmask[mt][r] = (e < E) ? 1.0f : 0.0f;
        e = e < E ? e : E - 1;
        obase[mt][r] = dstIdx[e] * FDIM + n;
      }

    // ---- stage 2+3 fused: W = ssp(T1@w2^T+b2); msg = (xj@fc1^T)*W; scatter ----
    for (int t = 0; t < 8; ++t) {
      v8f acc1[MT] = {};
      v8f acc2[MT] = {};
      #pragma unroll
      for (int kc = 0; kc < 4; ++kc) {
        v16h bw2 = load_b_lds(sW2, t, kc, n, sel);
        #pragma unroll
        for (int mt = 0; mt < MT; ++mt)
          acc1[mt] = __builtin_amdgcn_wmma_f32_16x16x32_f16(
              false, At1[mt][kc], false, bw2, (short)0, acc1[mt], false, false);
        v16h bfc = load_b_lds(sFC, t, kc, n, sel);
        #pragma unroll
        for (int mt = 0; mt < MT; ++mt)
          acc2[mt] = __builtin_amdgcn_wmma_f32_16x16x32_f16(
              false, Axj[mt][kc], false, bfc, (short)0, acc2[mt], false, false);
      }
      float bias2 = sB2[t * 16 + n];
      #pragma unroll
      for (int mt = 0; mt < MT; ++mt)
        #pragma unroll
        for (int r = 0; r < 8; ++r) {
          float Wv  = ssp(acc1[mt][r] + bias2);
          float msg = acc2[mt][r] * Wv * vmask[mt][r];
          unsafeAtomicAdd(out + obase[mt][r] + t * 16, msg);
        }
    }
  }
}

extern "C" void kernel_launch(void* const* d_in, const int* in_sizes, int n_in,
                              void* d_out, int out_size, void* d_ws, size_t ws_size,
                              hipStream_t stream) {
  const float* hfeat = (const float*)d_in[0];
  const int*   ei    = (const int*)d_in[1];
  const float* ea    = (const float*)d_in[2];
  const float* fc1   = (const float*)d_in[3];
  const float* w1    = (const float*)d_in[4];
  const float* b1    = (const float*)d_in[5];
  const float* w2    = (const float*)d_in[6];
  const float* b2    = (const float*)d_in[7];
  float* out = (float*)d_out;

  const int E = in_sizes[2] / HDIM;     // edge_attr rows

  // out is accumulated atomically -> zero it first (harness poisons d_out)
  const int n4 = out_size / 4;
  zero_kernel<<<(n4 + 255) / 256, 256, 0, stream>>>(out, n4);

  const int nTiles = (E + MT * 16 - 1) / (MT * 16);
  int blocks = 640;
  if (blocks * WAVES > nTiles) blocks = (nTiles + WAVES - 1) / WAVES;
  if (blocks < 1) blocks = 1;

  const size_t shmem = (size_t)3 * HDIM * FDIM * sizeof(_Float16)   // weights f16
                     + (size_t)2 * FDIM * sizeof(float)             // biases
                     + (size_t)WAVES * MT * 16 * HDIM * sizeof(_Float16); // T1
  cfconv_kernel<<<blocks, 32 * WAVES, shmem, stream>>>(
      hfeat, ei, ea, fc1, w1, b1, w2, b2, out, E);
}
